// ViGNN_72241349918943
// MI455X (gfx1250) — matrix operand order repaired and checked by reading, not compiled, not run on hardware
//
#include <hip/hip_runtime.h>
#include <hip/hip_bf16.h>

typedef __attribute__((ext_vector_type(16))) __bf16 v16bf;
typedef __attribute__((ext_vector_type(8)))  float  v8f;
typedef __attribute__((ext_vector_type(4)))  float  f4;

#define KNN_K 9
#define NEG_INF (-3.402823466e38f)

__device__ __forceinline__ float gelu_exact(float x) {
  return 0.5f * x * (1.0f + erff(x * 0.70710678118654752f));
}
__device__ __forceinline__ float bn_apply(float v, float g, float b, float m, float var) {
  return (v - m) * (g * rsqrtf(var + 1e-5f)) + b;
}

// ---------------------------------------------------------------------------
// WMMA bf16 GEMM: Y[M,N] = epi( X[M,K] * W[N,ldw](cols koff..koff+K [- koff2..])^T )
// Requirements (guaranteed by call sites): K % 16 == 0; X rows, W rows, koff(2)
// all 16-byte aligned. M/N overhang handled by index clamping (stores guarded).
// Block = 128 threads (4 waves). Wave w computes rows [bx*64+16w, +16) x
// cols [by*64, +64) as 4 WMMA tiles sharing one A fragment.
// ---------------------------------------------------------------------------
__global__ void gemm_wmma_bf16(
    const float* __restrict__ X, int M, int K, int ldx,
    const float* __restrict__ Wt, int ldw, int koff, int koff2,
    const float* __restrict__ bias,
    const float* __restrict__ bng, const float* __restrict__ bnb,
    const float* __restrict__ bnm, const float* __restrict__ bnv,
    const float* __restrict__ res,
    float* __restrict__ Y, int N, int act)
{
  int lane = threadIdx.x & 31;
  int wave = threadIdx.x >> 5;
  int half = lane >> 4;
  int l15  = lane & 15;
  int m0 = (blockIdx.x * 4 + wave) * 16;
  int n0 = blockIdx.y * 64;

  int arow = m0 + l15; if (arow >= M) arow = M - 1;
  const float* aptr = X + (size_t)arow * ldx;

  // clamped B column per sub-tile
  int bc[4];
#pragma unroll
  for (int t = 0; t < 4; ++t) {
    int c = n0 + t * 16 + l15;
    bc[t] = (c < N) ? c : (N - 1);
  }

  v8f z = {};
  v8f acc[4] = {z, z, z, z};

  int K32 = K & ~31;
  for (int k0 = 0; k0 < K32; k0 += 32) {
    // A 16x32 fragment: lane row = l15; K = {8*half..+7, 16+8*half..+7}
    v16bf a;
    {
      const float* ap = aptr + k0 + (half << 3);
      f4 a0 = *(const f4*)(ap);
      f4 a1 = *(const f4*)(ap + 4);
      f4 a2 = *(const f4*)(ap + 16);
      f4 a3 = *(const f4*)(ap + 20);
#pragma unroll
      for (int i = 0; i < 4; ++i) {
        a[i]      = (__bf16)a0[i];
        a[4 + i]  = (__bf16)a1[i];
        a[8 + i]  = (__bf16)a2[i];
        a[12 + i] = (__bf16)a3[i];
      }
    }
#pragma unroll
    for (int t = 0; t < 4; ++t) {
      // B 32x16 fragment: lane col = l15; K = {8*half..+7, 16+8*half..+7}
      const float* bp = Wt + (size_t)bc[t] * ldw + koff + k0 + (half << 3);
      f4 b0 = *(const f4*)(bp);
      f4 b1 = *(const f4*)(bp + 4);
      f4 b2 = *(const f4*)(bp + 16);
      f4 b3 = *(const f4*)(bp + 20);
      if (koff2 >= 0) {
        const float* bq = Wt + (size_t)bc[t] * ldw + koff2 + k0 + (half << 3);
        b0 -= *(const f4*)(bq);
        b1 -= *(const f4*)(bq + 4);
        b2 -= *(const f4*)(bq + 16);
        b3 -= *(const f4*)(bq + 20);
      }
      v16bf b;
#pragma unroll
      for (int i = 0; i < 4; ++i) {
        b[i]      = (__bf16)b0[i];
        b[4 + i]  = (__bf16)b1[i];
        b[8 + i]  = (__bf16)b2[i];
        b[12 + i] = (__bf16)b3[i];
      }
      acc[t] = __builtin_amdgcn_wmma_f32_16x16x32_bf16(false, a, false, b,
                                                       (short)0, acc[t], false, false);
    }
  }

  if (K32 < K) {           // 16-wide tail (K % 16 == 0 at every call site)
    int k0 = K32;
    v16bf a;
    {
      const float* ap = aptr + k0 + (half << 3);   // K = 8*half..+7 valid
      f4 a0 = *(const f4*)(ap);
      f4 a1 = *(const f4*)(ap + 4);
#pragma unroll
      for (int i = 0; i < 4; ++i) {
        a[i]      = (__bf16)a0[i];
        a[4 + i]  = (__bf16)a1[i];
        a[8 + i]  = (__bf16)0.0f;                  // K >= k0+16: zero
        a[12 + i] = (__bf16)0.0f;
      }
    }
#pragma unroll
    for (int t = 0; t < 4; ++t) {
      const float* bp = Wt + (size_t)bc[t] * ldw + koff + k0 + (half << 3);  // K = 8*half..+7 valid
      f4 b0 = *(const f4*)(bp);
      f4 b1 = *(const f4*)(bp + 4);
      if (koff2 >= 0) {
        const float* bq = Wt + (size_t)bc[t] * ldw + koff2 + k0 + (half << 3);
        b0 -= *(const f4*)(bq);
        b1 -= *(const f4*)(bq + 4);
      }
      v16bf b;
#pragma unroll
      for (int i = 0; i < 4; ++i) {
        b[i]      = (__bf16)b0[i];
        b[4 + i]  = (__bf16)b1[i];
        b[8 + i]  = (__bf16)0.0f;                  // K >= k0+16: zero
        b[12 + i] = (__bf16)0.0f;
      }
      acc[t] = __builtin_amdgcn_wmma_f32_16x16x32_bf16(false, a, false, b,
                                                       (short)0, acc[t], false, false);
    }
  }

  // epilogue
#pragma unroll
  for (int t = 0; t < 4; ++t) {
    int n = n0 + t * 16 + l15;
    if (n >= N) continue;
    float bz = bias ? bias[n] : 0.0f;
    float sc = 1.0f, sh = 0.0f;
    if (bng) {
      sc = bng[n] * rsqrtf(bnv[n] + 1e-5f);
      sh = bnb[n] - bnm[n] * sc;
    }
    v8f av = acc[t];
#pragma unroll
    for (int r = 0; r < 8; ++r) {
      int m = m0 + r + (half << 3);   // C/D layout: VGPR r -> row r + 8*half
      if (m >= M) continue;
      float v = av[r] + bz;
      v = v * sc + sh;
      if (res) v += res[(size_t)m * N + n];
      if (act == 1) v = gelu_exact(v);
      Y[(size_t)m * N + n] = v;
    }
  }
}

// ---------------------------------------------------------------------------
// Direct 3x3 conv, NCHW input, pad=1, fused BN (+ optional ReLU),
// optional NHWC ([B, Ho*Wo, Co]) output for handoff to the GEMM pipeline.
// ---------------------------------------------------------------------------
__global__ void conv3x3_nchw(const float* __restrict__ x,
    const float* __restrict__ w, const float* __restrict__ bias,
    const float* __restrict__ bng, const float* __restrict__ bnb,
    const float* __restrict__ bnm, const float* __restrict__ bnv,
    float* __restrict__ y,
    int B, int Ci, int H, int W, int Co, int Ho, int Wo,
    int stride, int relu, int nhwc_out)
{
  int t = blockIdx.x * blockDim.x + threadIdx.x;
  int total = B * Co * Ho * Wo;
  if (t >= total) return;
  int wo = t % Wo; int tmp = t / Wo;
  int ho = tmp % Ho; tmp /= Ho;
  int co = tmp % Co; int b = tmp / Co;
  float acc = bias[co];
  for (int ci = 0; ci < Ci; ++ci)
    for (int kh = 0; kh < 3; ++kh) {
      int h = ho * stride + kh - 1;
      if ((unsigned)h >= (unsigned)H) continue;
      for (int kw = 0; kw < 3; ++kw) {
        int ww = wo * stride + kw - 1;
        if ((unsigned)ww >= (unsigned)W) continue;
        acc += x[(((size_t)b * Ci + ci) * H + h) * W + ww] *
               w[(((size_t)co * Ci + ci) * 3 + kh) * 3 + kw];
      }
    }
  acc = bn_apply(acc, bng[co], bnb[co], bnm[co], bnv[co]);
  if (relu) acc = fmaxf(acc, 0.0f);
  if (nhwc_out)
    y[(((size_t)b * Ho + ho) * Wo + wo) * Co + co] = acc;
  else
    y[(((size_t)b * Co + co) * Ho + ho) * Wo + wo] = acc;
}

// Downsample 3x3 stride-2 conv on NHWC activations, fused BN. NHWC output.
__global__ void conv3x3_nhwc(const float* __restrict__ x,
    const float* __restrict__ w, const float* __restrict__ bias,
    const float* __restrict__ bng, const float* __restrict__ bnb,
    const float* __restrict__ bnm, const float* __restrict__ bnv,
    float* __restrict__ y,
    int B, int Ci, int H, int W, int Co, int Ho, int Wo)
{
  int t = blockIdx.x * blockDim.x + threadIdx.x;
  int total = B * Co * Ho * Wo;
  if (t >= total) return;
  int co = t % Co; int tmp = t / Co;
  int wo = tmp % Wo; tmp /= Wo;
  int ho = tmp % Ho; int b = tmp / Ho;
  float acc = bias[co];
  for (int kh = 0; kh < 3; ++kh) {
    int h = ho * 2 + kh - 1;
    if ((unsigned)h >= (unsigned)H) continue;
    for (int kw = 0; kw < 3; ++kw) {
      int ww = wo * 2 + kw - 1;
      if ((unsigned)ww >= (unsigned)W) continue;
      const float* xp = x + (((size_t)b * H + h) * W + ww) * Ci;
      const float* wp = w + (size_t)co * Ci * 9 + kh * 3 + kw;
      for (int ci = 0; ci < Ci; ++ci)
        acc += xp[ci] * wp[(size_t)ci * 9];
    }
  }
  acc = bn_apply(acc, bng[co], bnb[co], bnm[co], bnv[co]);
  y[(((size_t)b * Ho + ho) * Wo + wo) * Co + co] = acc;
}

// F[r,:] = T[r,:] / max(||T[r,:]||, 1e-12)   (C % 4 == 0)
__global__ void normalize_rows(const float* __restrict__ T, float* __restrict__ F,
                               int rows, int C)
{
  int r = blockIdx.x * blockDim.x + threadIdx.x;
  if (r >= rows) return;
  const float* src = T + (size_t)r * C;
  float s = 0.0f;
  for (int c = 0; c < C; c += 4) {
    f4 v = *(const f4*)(src + c);
    s += v[0]*v[0] + v[1]*v[1] + v[2]*v[2] + v[3]*v[3];
  }
  float inv = 1.0f / fmaxf(sqrtf(s), 1e-12f);
  float* dst = F + (size_t)r * C;
  for (int c = 0; c < C; c += 4) {
    f4 v = *(const f4*)(src + c);
    v *= inv;
    *(f4*)(dst + c) = v;
  }
}

// KNN: rows are unit vectors, so argmin dist == argmax dot (tie -> lowest index,
// matching jax top_k). One 256-thread block per query row; 9 argmax rounds.
__global__ void knn_topk(const float* __restrict__ F, int* __restrict__ idxOut,
                         int N, int C)
{
  __shared__ float fi[384];
  __shared__ float rd[256];
  __shared__ int   ri[256];
  int row = blockIdx.x;            // b*N + i
  int b   = row / N;
  int tid = threadIdx.x;
  const float* Fb = F + (size_t)b * N * C;
  const float* Fi = F + (size_t)row * C;
  for (int c = tid; c < C; c += 256) fi[c] = Fi[c];
  __syncthreads();

  float bd[KNN_K]; int bi[KNN_K];
#pragma unroll
  for (int s = 0; s < KNN_K; ++s) { bd[s] = NEG_INF; bi[s] = 0x7fffffff; }

  for (int j = tid; j < N; j += 256) {
    const float* Fj = Fb + (size_t)j * C;
    float d = 0.0f;
    for (int c = 0; c < C; c += 4) {
      f4 v = *(const f4*)(Fj + c);
      d += fi[c] * v[0] + fi[c+1] * v[1] + fi[c+2] * v[2] + fi[c+3] * v[3];
    }
    if (d > bd[KNN_K - 1] || (d == bd[KNN_K - 1] && j < bi[KNN_K - 1])) {
      int p = KNN_K - 1;
      while (p > 0 && (d > bd[p - 1] || (d == bd[p - 1] && j < bi[p - 1]))) {
        bd[p] = bd[p - 1]; bi[p] = bi[p - 1]; --p;
      }
      bd[p] = d; bi[p] = j;
    }
  }

  for (int round = 0; round < KNN_K; ++round) {
    float d = NEG_INF; int ix = 0x7fffffff;
#pragma unroll
    for (int s = 0; s < KNN_K; ++s)
      if (bd[s] > d || (bd[s] == d && bi[s] < ix)) { d = bd[s]; ix = bi[s]; }
    rd[tid] = d; ri[tid] = ix;
    __syncthreads();
    for (int off = 128; off > 0; off >>= 1) {
      if (tid < off) {
        float d2 = rd[tid + off]; int i2 = ri[tid + off];
        if (d2 > rd[tid] || (d2 == rd[tid] && i2 < ri[tid])) {
          rd[tid] = d2; ri[tid] = i2;
        }
      }
      __syncthreads();
    }
    int w = ri[0];
    __syncthreads();
    if (tid == 0) idxOut[(size_t)row * KNN_K + round] = w;
#pragma unroll
    for (int s = 0; s < KNN_K; ++s)
      if (bi[s] == w) { bd[s] = NEG_INF; bi[s] = 0x7fffffff; }
  }
}

// edge-conv epilogue: G = gelu(bn( U[p,o] + max_k V[b, idx[p,k], o] + bias[o] ))
__global__ void edge_max_bn_gelu(const float* __restrict__ U,
    const float* __restrict__ V, const int* __restrict__ idx,
    const float* __restrict__ bias,
    const float* __restrict__ g, const float* __restrict__ bb,
    const float* __restrict__ m, const float* __restrict__ var,
    float* __restrict__ G, int M, int N, int C2)
{
  int t = blockIdx.x * blockDim.x + threadIdx.x;
  if (t >= M * C2) return;
  int o = t % C2; int pnt = t / C2;
  int b = pnt / N;
  const int* ip = idx + (size_t)pnt * KNN_K;
  size_t base = (size_t)b * N * C2;
  float best = NEG_INF;
  for (int k = 0; k < KNN_K; ++k) {
    int j = ip[k];
    best = fmaxf(best, V[base + (size_t)j * C2 + o]);
  }
  float v2 = U[(size_t)pnt * C2 + o] + best + bias[o];
  v2 = bn_apply(v2, g[o], bb[o], m[o], var[o]);
  G[(size_t)pnt * C2 + o] = gelu_exact(v2);
}

// P[b,c] = bn( mean_n X[b,n,c] )
__global__ void pool_bn(const float* __restrict__ X, float* __restrict__ P,
    int B, int N, int C, const float* __restrict__ g, const float* __restrict__ bb,
    const float* __restrict__ m, const float* __restrict__ v)
{
  int t = blockIdx.x * blockDim.x + threadIdx.x;
  if (t >= B * C) return;
  int c = t % C, b = t / C;
  float s = 0.0f;
  for (int n = 0; n < N; ++n) s += X[((size_t)b * N + n) * C + c];
  s /= (float)N;
  P[t] = bn_apply(s, g[c], bb[c], m[c], v[c]);
}

// ---------------------------------------------------------------------------
// Host orchestration
// ---------------------------------------------------------------------------
struct StageP {
  const float *fc1w,*fc1b,*fc1g,*fc1bb,*fc1m,*fc1v;
  const float *gcw,*gcb,*gcg,*gcbb,*gcm,*gcv;
  const float *fc2w,*fc2b,*fc2g,*fc2bb,*fc2m,*fc2v;
  const float *ffc1w,*ffc1b,*ffc1g,*ffc1bb,*ffc1m,*ffc1v;
  const float *ffc2w,*ffc2b,*ffc2g,*ffc2bb,*ffc2m,*ffc2v;
};
struct DownP { const float *w,*b,*g,*bb,*m,*v; };

static inline int cdiv(int a, int b) { return (a + b - 1) / b; }

extern "C" void kernel_launch(void* const* d_in, const int* in_sizes, int n_in,
                              void* d_out, int out_size, void* d_ws, size_t ws_size,
                              hipStream_t stream)
{
  (void)in_sizes; (void)n_in; (void)out_size;
  int ip_ = 0;
  auto nf = [&]() -> const float* { return (const float*)d_in[ip_++]; };

  const float* x = nf();
  // stem
  const float *c1w=nf(), *c1b=nf(), *bn1g=nf(), *bn1b=nf(), *bn1m=nf(), *bn1v=nf();
  const float *c2w=nf(), *c2b=nf(), *bn2g=nf(), *bn2b=nf(), *bn2m=nf(), *bn2v=nf();
  const float *c3w=nf(), *c3b=nf(), *bn3g=nf(), *bn3b=nf(), *bn3m=nf(), *bn3v=nf();
  StageP st[4];
  for (int s = 0; s < 4; ++s) {
    st[s].fc1w=nf();  st[s].fc1b=nf();  st[s].fc1g=nf();  st[s].fc1bb=nf();  st[s].fc1m=nf();  st[s].fc1v=nf();
    st[s].gcw=nf();   st[s].gcb=nf();   st[s].gcg=nf();   st[s].gcbb=nf();   st[s].gcm=nf();   st[s].gcv=nf();
    st[s].fc2w=nf();  st[s].fc2b=nf();  st[s].fc2g=nf();  st[s].fc2bb=nf();  st[s].fc2m=nf();  st[s].fc2v=nf();
    st[s].ffc1w=nf(); st[s].ffc1b=nf(); st[s].ffc1g=nf(); st[s].ffc1bb=nf(); st[s].ffc1m=nf(); st[s].ffc1v=nf();
    st[s].ffc2w=nf(); st[s].ffc2b=nf(); st[s].ffc2g=nf(); st[s].ffc2bb=nf(); st[s].ffc2m=nf(); st[s].ffc2v=nf();
  }
  DownP dn[3];
  for (int d = 0; d < 3; ++d) {
    dn[d].w=nf(); dn[d].b=nf(); dn[d].g=nf(); dn[d].bb=nf(); dn[d].m=nf(); dn[d].v=nf();
  }
  const float *normg=nf(), *normbb=nf(), *normm=nf(), *normv=nf();
  const float *headw=nf(), *headb=nf();

  // workspace carve-up (floats); total ~8.55M floats ~= 33 MB
  float* ws = (float*)d_ws;
  size_t o = 0;
  float* XA  = ws + o; o += 602112;    // [B*3136, 48] max activation
  float* XB  = ws + o; o += 602112;
  float* T   = ws + o; o += 602112;    // post g_fc1+bn
  float* F   = ws + o; o += 602112;    // normalized features
  float* U   = ws + o; o += 1204224;   // T*(w1-w2)^T  [M, 2C]
  float* V   = ws + o; o += 1204224;   // T*w2^T       [M, 2C]
  float* G   = ws + o; o += 1204224;   // edge-conv output
  float* MID = ws + o; o += 2408448;   // FFN hidden   [M, 4C]
  float* P   = ws + o; o += 1536;      // pooled [B, 384]
  int*   IDX = (int*)(ws + o); o += 112896;
  if (ws_size < o * sizeof(float)) return;
  float* S1 = U;   // stem scratch aliases (stem completes before stages)
  float* S2 = V;

  const int B = 4;
  const int chans[4] = {48, 96, 240, 384};

  // ---- stem ----
  {
    int tot = B * 24 * 112 * 112;
    conv3x3_nchw<<<cdiv(tot, 256), 256, 0, stream>>>(
        x, c1w, c1b, bn1g, bn1b, bn1m, bn1v, S1,
        B, 3, 224, 224, 24, 112, 112, 2, 1, 0);
    tot = B * 48 * 56 * 56;
    conv3x3_nchw<<<cdiv(tot, 256), 256, 0, stream>>>(
        S1, c2w, c2b, bn2g, bn2b, bn2m, bn2v, S2,
        B, 24, 112, 112, 48, 56, 56, 2, 1, 0);
    conv3x3_nchw<<<cdiv(tot, 256), 256, 0, stream>>>(
        S2, c3w, c3b, bn3g, bn3b, bn3m, bn3v, XA,
        B, 48, 56, 56, 48, 56, 56, 1, 0, 1);   // -> NHWC point layout
  }

  float* cur = XA;
  float* oth = XB;
  int Hh = 56;
  for (int s = 0; s < 4; ++s) {
    int C = chans[s];
    int N = Hh * Hh;
    int M = B * N;
    int C2 = 2 * C, C4 = 4 * C;
    dim3 blk(128);

    // t = bn(g_fc1(x))
    gemm_wmma_bf16<<<dim3(cdiv(M,64), cdiv(C,64)), blk, 0, stream>>>(
        cur, M, C, C, st[s].fc1w, C, 0, -1, st[s].fc1b,
        st[s].fc1g, st[s].fc1bb, st[s].fc1m, st[s].fc1v, nullptr, T, C, 0);
    // knn on normalized t
    normalize_rows<<<cdiv(M, 256), 256, 0, stream>>>(T, F, M, C);
    knn_topk<<<M, 256, 0, stream>>>(F, IDX, N, C);
    // edge conv as two GEMMs: U = T*(w1-w2)^T, V = T*w2^T
    gemm_wmma_bf16<<<dim3(cdiv(M,64), cdiv(C2,64)), blk, 0, stream>>>(
        T, M, C, C, st[s].gcw, C2, 0, C, nullptr,
        nullptr, nullptr, nullptr, nullptr, nullptr, U, C2, 0);
    gemm_wmma_bf16<<<dim3(cdiv(M,64), cdiv(C2,64)), blk, 0, stream>>>(
        T, M, C, C, st[s].gcw, C2, C, -1, nullptr,
        nullptr, nullptr, nullptr, nullptr, nullptr, V, C2, 0);
    edge_max_bn_gelu<<<cdiv(M * C2, 256), 256, 0, stream>>>(
        U, V, IDX, st[s].gcb, st[s].gcg, st[s].gcbb, st[s].gcm, st[s].gcv,
        G, M, N, C2);
    // t = bn(g_fc2(G)) + x   -> oth
    gemm_wmma_bf16<<<dim3(cdiv(M,64), cdiv(C,64)), blk, 0, stream>>>(
        G, M, C2, C2, st[s].fc2w, C2, 0, -1, st[s].fc2b,
        st[s].fc2g, st[s].fc2bb, st[s].fc2m, st[s].fc2v, cur, oth, C, 0);
    // FFN: mid = gelu(bn(f_fc1(t)))
    gemm_wmma_bf16<<<dim3(cdiv(M,64), cdiv(C4,64)), blk, 0, stream>>>(
        oth, M, C, C, st[s].ffc1w, C, 0, -1, st[s].ffc1b,
        st[s].ffc1g, st[s].ffc1bb, st[s].ffc1m, st[s].ffc1v, nullptr, MID, C4, 1);
    // x = bn(f_fc2(mid)) + t  -> cur
    gemm_wmma_bf16<<<dim3(cdiv(M,64), cdiv(C,64)), blk, 0, stream>>>(
        MID, M, C4, C4, st[s].ffc2w, C4, 0, -1, st[s].ffc2b,
        st[s].ffc2g, st[s].ffc2bb, st[s].ffc2m, st[s].ffc2v, oth, cur, C, 0);

    if (s < 3) {
      int Co = chans[s + 1], Ho = Hh / 2;
      int tot = B * Ho * Ho * Co;
      conv3x3_nhwc<<<cdiv(tot, 256), 256, 0, stream>>>(
          cur, dn[s].w, dn[s].b, dn[s].g, dn[s].bb, dn[s].m, dn[s].v, oth,
          B, C, Hh, Hh, Co, Ho, Ho);
      float* t2 = cur; cur = oth; oth = t2;
      Hh = Ho;
    }
  }

  // mean pool + final BN
  pool_bn<<<cdiv(B * 384, 256), 256, 0, stream>>>(
      cur, P, B, Hh * Hh, 384, normg, normbb, normm, normv);
  // head: [4,384] @ [1000,384]^T + b -> d_out
  gemm_wmma_bf16<<<dim3(1, cdiv(1000, 64)), dim3(128), 0, stream>>>(
      P, B, 384, 384, headw, 384, 0, -1, headb,
      nullptr, nullptr, nullptr, nullptr, nullptr, (float*)d_out, 1000, 0);
}
